// DistillerGRU_27401891348885
// MI455X (gfx1250) — matrix operand order repaired and checked by reading, not compiled, not run on hardware
//
#include <hip/hip_runtime.h>
#include <hip/hip_bf16.h>

#define T_  512
#define B_  64
#define H_  1024
#define H3_ 3072

typedef __attribute__((ext_vector_type(8)))  float  v8f;
typedef __attribute__((ext_vector_type(8)))  __bf16 v8bf;
typedef __attribute__((ext_vector_type(16))) __bf16 v16bf;

// ---------------------------------------------------------------------------
// Fragment loader: two 16-byte chunks -> one v16bf WMMA operand
// ---------------------------------------------------------------------------
static __device__ __forceinline__ v16bf load_frag(const __bf16* p0, const __bf16* p1) {
    v8bf lo = *(const v8bf*)p0;
    v8bf hi = *(const v8bf*)p1;
    v16bf r;
#pragma unroll
    for (int i = 0; i < 8; ++i) { r[i] = lo[i]; r[i + 8] = hi[i]; }
    return r;
}

static __device__ __forceinline__ v8f wmma_bf16(v16bf a, v16bf b, v8f c) {
    return __builtin_amdgcn_wmma_f32_16x16x32_bf16(false, a, false, b, (short)0, c,
                                                   false, false);
}

static __device__ __forceinline__ float sigmoidf_(float x) {
    return 1.0f / (1.0f + __expf(-x));
}

// ---------------------------------------------------------------------------
// Grid-wide spin barrier (persistent-kernel style; <=32 blocks co-resident)
// ---------------------------------------------------------------------------
static __device__ __forceinline__ void grid_sync(unsigned* arrive, unsigned* phase,
                                                 unsigned nblk, unsigned cur) {
    __syncthreads();
    if (threadIdx.x == 0) {
        __threadfence();
        unsigned prev = atomicAdd(arrive, 1u);
        if (prev == nblk - 1u) {
            atomicExch(arrive, 0u);
            __threadfence();
            atomicAdd(phase, 1u);
        } else {
            volatile unsigned* vp = phase;
            while (*vp == cur) { __builtin_amdgcn_s_sleep(2); }
        }
    }
    __syncthreads();
}

// ---------------------------------------------------------------------------
// f32 -> bf16 elementwise convert
// ---------------------------------------------------------------------------
__global__ void __launch_bounds__(256) cvt_bf16_kernel(const float* __restrict__ src,
                                                       __bf16* __restrict__ dst, int n) {
    for (int i = blockIdx.x * blockDim.x + threadIdx.x; i < n; i += gridDim.x * blockDim.x)
        dst[i] = (__bf16)src[i];
}

// ---------------------------------------------------------------------------
// gx = A[M,K=H] x W[N=3H,K=H]^T + bias
// Register tiling: each wave computes a 32x64 output block (2 M-tiles x
// 4 N-tiles): 2 A-frags + 4 B-frags feed 8 WMMAs per 32-wide k-step.
// ---------------------------------------------------------------------------
__global__ void __launch_bounds__(256) gemm_bias_kernel(const __bf16* __restrict__ A,
                                                        const __bf16* __restrict__ W,
                                                        const float*  __restrict__ bias,
                                                        float* __restrict__ C,
                                                        int mtiles2) {            // M/32
    const int lane = threadIdx.x & 31;
    const int wave = blockIdx.x * (blockDim.x >> 5) + (threadIdx.x >> 5);
    const int NT4  = H3_ / 64;                 // 48 N-groups of 4 tiles
    const int tn4  = wave % NT4;
    const int tm2  = wave / NT4;
    if (tm2 >= mtiles2) return;                // wave-uniform; EXEC stays all-1 inside

    const int half = lane >> 4;
    const int l15  = lane & 15;
    const __bf16* arow0 = A + (size_t)(tm2 * 32 + l15) * H_ + half * 16;
    const __bf16* arow1 = arow0 + (size_t)16 * H_;
    const __bf16* wrow0 = W + (size_t)(tn4 * 64 + l15) * H_ + half * 16;

    v8f acc[2][4] = {};
#pragma unroll 2
    for (int kk = 0; kk < H_; kk += 32) {
        __builtin_prefetch(arow0 + kk + 128, 0, 1);
        v16bf a0 = load_frag(arow0 + kk, arow0 + kk + 8);
        v16bf a1 = load_frag(arow1 + kk, arow1 + kk + 8);
#pragma unroll
        for (int n = 0; n < 4; ++n) {
            const __bf16* wr = wrow0 + (size_t)(n * 16) * H_;
            v16bf b = load_frag(wr + kk, wr + kk + 8);
            acc[0][n] = wmma_bf16(a0, b, acc[0][n]);
            acc[1][n] = wmma_bf16(a1, b, acc[1][n]);
        }
    }
#pragma unroll
    for (int n = 0; n < 4; ++n) {
        const int col = tn4 * 64 + n * 16 + l15;
        const float bi = bias[col];
#pragma unroll
        for (int mi = 0; mi < 2; ++mi) {
#pragma unroll
            for (int i = 0; i < 8; ++i) {
                int row = tm2 * 32 + mi * 16 + i + half * 8;
                C[(size_t)row * H3_ + col] = acc[mi][n][i] + bi;
            }
        }
    }
}

// ---------------------------------------------------------------------------
// Persistent GRU scan for one layer. 16 blocks x 256 thr = 128 waves =
// 2 batch-pair-tiles x 64 h-tiles. Each wave computes r/z/n gh-tiles for
// TWO 16-row batch tiles (Wh frags reused), applies the gate math, and
// double-buffers h.
// ---------------------------------------------------------------------------
__global__ void __launch_bounds__(256) gru_scan_kernel(const float*  __restrict__ gx,    // [T,B,3H]
                                                       const float*  __restrict__ bh,    // [3H]
                                                       const __bf16* __restrict__ Wh,    // [3H,H] bf16
                                                       const float*  __restrict__ h0,    // [B,H]
                                                       float*  __restrict__ hbuf,        // [2,B,H]
                                                       __bf16* __restrict__ hbufb,       // [2,B,H]
                                                       float*  __restrict__ yout,        // [T,B,H]
                                                       __bf16* __restrict__ youtb,       // [T,B,H] or null
                                                       float*  __restrict__ hfin,        // [B,H]
                                                       unsigned* __restrict__ bar) {     // [2]
    const int tid  = blockIdx.x * blockDim.x + threadIdx.x;
    const int lane = threadIdx.x & 31;
    const int wave = tid >> 5;      // 0..127
    const int mg   = wave & 1;      // batch pair-tile: rows [mg*32, mg*32+32)
    const int j    = wave >> 1;     // h tile 0..63
    const int half = lane >> 4;
    const int l15  = lane & 15;
    const int nthr = gridDim.x * blockDim.x;
    unsigned phase = 0;

    // stage initial hidden state into the double buffer (slot 0)
    for (int i = tid; i < B_ * H_; i += nthr) {
        float v  = h0[i];
        hbuf[i]  = v;
        hbufb[i] = (__bf16)v;
    }
    grid_sync(bar, bar + 1, gridDim.x, phase); ++phase;

    const int   gcol = j * 16 + l15;                 // output column within H
    const float br = bh[gcol], bz = bh[H_ + gcol], bn = bh[2 * H_ + gcol];
    const __bf16* wrr = Wh + (size_t)(gcol)          * H_ + half * 16;
    const __bf16* wrz = Wh + (size_t)(H_     + gcol) * H_ + half * 16;
    const __bf16* wrn = Wh + (size_t)(2 * H_ + gcol) * H_ + half * 16;

    for (int t = 0; t < T_; ++t) {
        const int cur = t & 1, nxt = cur ^ 1;
        const __bf16* hb0 = hbufb + (size_t)cur * B_ * H_
                                  + (size_t)(mg * 32 + l15) * H_ + half * 16;
        const __bf16* hb1 = hb0 + (size_t)16 * H_;
        v8f ar[2] = {}, az[2] = {}, an[2] = {};
#pragma unroll 2
        for (int kk = 0; kk < H_; kk += 32) {
            v16bf a0 = load_frag(hb0 + kk, hb0 + kk + 8);
            v16bf a1 = load_frag(hb1 + kk, hb1 + kk + 8);
            v16bf bR = load_frag(wrr + kk, wrr + kk + 8);
            ar[0] = wmma_bf16(a0, bR, ar[0]);
            ar[1] = wmma_bf16(a1, bR, ar[1]);
            v16bf bZ = load_frag(wrz + kk, wrz + kk + 8);
            az[0] = wmma_bf16(a0, bZ, az[0]);
            az[1] = wmma_bf16(a1, bZ, az[1]);
            v16bf bN = load_frag(wrn + kk, wrn + kk + 8);
            an[0] = wmma_bf16(a0, bN, an[0]);
            an[1] = wmma_bf16(a1, bN, an[1]);
        }

        const float* gxt  = gx   + (size_t)t   * B_ * H3_;
        const float* hcur = hbuf + (size_t)cur * B_ * H_;
        float*  hnx  = hbuf  + (size_t)nxt * B_ * H_;
        __bf16* hnxb = hbufb + (size_t)nxt * B_ * H_;
        float*  yt   = yout  + (size_t)t   * B_ * H_;

#pragma unroll
        for (int mi = 0; mi < 2; ++mi) {
#pragma unroll
            for (int i = 0; i < 8; ++i) {
                const int    b  = mg * 32 + mi * 16 + i + half * 8;
                const size_t g3 = (size_t)b * H3_;
                const float rx = gxt[g3 + gcol];
                const float zx = gxt[g3 + H_ + gcol];
                const float nx = gxt[g3 + 2 * H_ + gcol];
                const float r  = sigmoidf_(rx + ar[mi][i] + br);
                const float z  = sigmoidf_(zx + az[mi][i] + bz);
                const float n  = tanhf(nx + r * (an[mi][i] + bn));
                const size_t hi = (size_t)b * H_ + gcol;
                const float hold = hcur[hi];
                const float hnew = (1.0f - z) * n + z * hold;
                hnx[hi]  = hnew;
                hnxb[hi] = (__bf16)hnew;
                yt[hi]   = hnew;
                if (youtb) youtb[(size_t)t * B_ * H_ + hi] = (__bf16)hnew;
            }
        }
        grid_sync(bar, bar + 1, gridDim.x, phase); ++phase;
    }

    // T_ is even -> final state resides in slot 0
    const float* hf = hbuf + (size_t)(T_ & 1) * B_ * H_;
    for (int i = tid; i < B_ * H_; i += nthr) hfin[i] = hf[i];
}

// ---------------------------------------------------------------------------
// Host launcher
// ---------------------------------------------------------------------------
extern "C" void kernel_launch(void* const* d_in, const int* in_sizes, int n_in,
                              void* d_out, int out_size, void* d_ws, size_t ws_size,
                              hipStream_t stream) {
    (void)in_sizes; (void)n_in; (void)out_size; (void)ws_size;

    const float* x   = (const float*)d_in[0];
    const float* h0  = (const float*)d_in[1];
    const float* Wx0 = (const float*)d_in[2];
    const float* bx0 = (const float*)d_in[3];
    const float* Wh0 = (const float*)d_in[4];
    const float* bh0 = (const float*)d_in[5];
    const float* Wx1 = (const float*)d_in[6];
    const float* bx1 = (const float*)d_in[7];
    const float* Wh1 = (const float*)d_in[8];
    const float* bh1 = (const float*)d_in[9];
    float* out = (float*)d_out;

    const size_t TBH = (size_t)T_ * B_ * H_;
    char* ws = (char*)d_ws;
    size_t off = 0;
    auto carve = [&](size_t bytes) -> void* {
        void* p = ws + off;
        off += (bytes + 255) & ~(size_t)255;
        return p;
    };

    __bf16* xbf   = (__bf16*)carve(TBH * 2);
    float*  y0f   = (float*) carve(TBH * 4);
    __bf16* y0bf  = (__bf16*)carve(TBH * 2);
    float*  gx    = (float*) carve((size_t)T_ * B_ * H3_ * 4);
    __bf16* Wx0b  = (__bf16*)carve((size_t)H3_ * H_ * 2);
    __bf16* Wh0b  = (__bf16*)carve((size_t)H3_ * H_ * 2);
    __bf16* Wx1b  = (__bf16*)carve((size_t)H3_ * H_ * 2);
    __bf16* Wh1b  = (__bf16*)carve((size_t)H3_ * H_ * 2);
    float*  hbuf  = (float*) carve((size_t)2 * B_ * H_ * 4);
    __bf16* hbufb = (__bf16*)carve((size_t)2 * B_ * H_ * 2);
    unsigned* bar = (unsigned*)carve(256);   // [0..1] layer0, [2..3] layer1

    hipMemsetAsync(bar, 0, 256, stream);

    // bf16 conversions
    cvt_bf16_kernel<<<4096, 256, 0, stream>>>(x,   xbf,  (int)TBH);
    cvt_bf16_kernel<<<1024, 256, 0, stream>>>(Wx0, Wx0b, H3_ * H_);
    cvt_bf16_kernel<<<1024, 256, 0, stream>>>(Wh0, Wh0b, H3_ * H_);
    cvt_bf16_kernel<<<1024, 256, 0, stream>>>(Wx1, Wx1b, H3_ * H_);
    cvt_bf16_kernel<<<1024, 256, 0, stream>>>(Wh1, Wh1b, H3_ * H_);

    const int MTILES2 = (T_ * B_) / 32;                        // 1024
    const int GEMM_BLOCKS = MTILES2 * (H3_ / 64) / 8;          // 6144 (8 waves/block)

    // ---- Layer 0 ----
    gemm_bias_kernel<<<GEMM_BLOCKS, 256, 0, stream>>>(xbf, Wx0b, bx0, gx, MTILES2);
    gru_scan_kernel<<<16, 256, 0, stream>>>(gx, bh0, Wh0b, h0,
                                            hbuf, hbufb, y0f, y0bf,
                                            out + TBH,            // h_final[0]
                                            bar);

    // ---- Layer 1 ----
    gemm_bias_kernel<<<GEMM_BLOCKS, 256, 0, stream>>>(y0bf, Wx1b, bx1, gx, MTILES2);
    gru_scan_kernel<<<16, 256, 0, stream>>>(gx, bh1, Wh1b, h0 + (size_t)B_ * H_,
                                            hbuf, hbufb, out /*y1*/, nullptr,
                                            out + TBH + (size_t)B_ * H_,  // h_final[1]
                                            bar + 2);
}
